// Encoder_90348932038693
// MI455X (gfx1250) — compile-verified, matrix operand order checked by reading, not run on hardware
//
#include <hip/hip_runtime.h>
#include <hip/hip_bf16.h>
#include <math.h>

typedef __attribute__((ext_vector_type(16))) __bf16    v16bf;
typedef __attribute__((ext_vector_type(2)))  __bf16    v2bf;
typedef __attribute__((ext_vector_type(8)))  float     v8f;
typedef __attribute__((ext_vector_type(2)))  float     v2f;
typedef __attribute__((ext_vector_type(8)))  unsigned  v8u;
typedef int v4i __attribute__((__vector_size__(16)));

#if __has_builtin(__builtin_amdgcn_global_load_async_to_lds_b128)
#define ASYNC_B 1
#else
#define ASYNC_B 0
#endif

__device__ inline void wait_async0() {
#if ASYNC_B
#if __has_builtin(__builtin_amdgcn_s_wait_asynccnt)
    __builtin_amdgcn_s_wait_asynccnt(0);
#else
    asm volatile("s_wait_asynccnt 0" ::: "memory");
#endif
#endif
}

// ---------- helpers ----------
__device__ inline unsigned pack2_bf16(float x, float y) {
#if __has_builtin(__builtin_amdgcn_cvt_pk_bf16_f32)
    v2bf v = __builtin_amdgcn_cvt_pk_bf16_f32(x, y);    // single packed convert
    return __builtin_bit_cast(unsigned, v);
#else
    v2bf v = {(__bf16)x, (__bf16)y};                    // native bf16 converts (RNE)
    return __builtin_bit_cast(unsigned, v);
#endif
}

// ---------- fp32 -> bf16 cast (weights) ----------
__global__ void k_cast_bf16(const float* __restrict__ s, unsigned short* __restrict__ d, long n) {
    long i = (long)blockIdx.x * 256 + threadIdx.x;
    if (i < n) {
        __bf16 v = (__bf16)s[i];
        d[i] = __builtin_bit_cast(unsigned short, v);
    }
}

// ---------- patch embed: 4x4/4 conv as per-token dot ----------
__global__ void k_conv(const float* __restrict__ x, const float* __restrict__ cw,
                       const float* __restrict__ cb, float* __restrict__ out, int B) {
    long idx = (long)blockIdx.x * 256 + threadIdx.x;
    long total = (long)B * 128 * 128 * 56;
    if (idx >= total) return;
    int c = (int)(idx % 56);
    long t = idx / 56;
    int w = (int)(t % 128);
    int h = (int)((t / 128) % 128);
    int b = (int)(t / (128 * 128));
    float s = cb[c];
    for (int ci = 0; ci < 3; ci++)
        for (int dh = 0; dh < 4; dh++)
            for (int dw = 0; dw < 4; dw++)
                s += x[(((long)b * 3 + ci) * 512 + (h * 4 + dh)) * 512 + (w * 4 + dw)] *
                     cw[((c * 3 + ci) * 4 + dh) * 4 + dw];
    out[t * 56 + c] = s;
}

// ---------- layernorm (one wave per token) ----------
__global__ __launch_bounds__(128) void k_ln(const float* __restrict__ in, float* __restrict__ out,
                                            const float* __restrict__ w, const float* __restrict__ b,
                                            long ntok, int C) {
    int wid = threadIdx.x >> 5, lane = threadIdx.x & 31;
    long t = (long)blockIdx.x * 4 + wid;
    if (t >= ntok) return;
    const float* row = in + t * C;
    float s = 0.f, sq = 0.f;
    for (int c = lane; c < C; c += 32) { float v = row[c]; s += v; sq += v * v; }
    for (int o = 16; o >= 1; o >>= 1) { s += __shfl_xor(s, o, 32); sq += __shfl_xor(sq, o, 32); }
    float mean = s / C;
    float var = sq / C - mean * mean;
    float rstd = rsqrtf(var + 1e-5f);
    float* orow = out + t * C;
    for (int c = lane; c < C; c += 32) orow[c] = (row[c] - mean) * rstd * w[c] + b[c];
}

// ---------- WMMA bf16 GEMM: out[M,N] = A[M,K] * W[N,K]^T + bias (+gelu) (+residual)
// Invariants from the launcher: M % 64 == 0, K % 8 == 0, all pointers 16B aligned.
#define BM 64
#define BN 64
#define BK 32
__global__ __launch_bounds__(256) void k_gemm(const float* __restrict__ A,
                                              const unsigned short* __restrict__ Wb,
                                              const float* __restrict__ bias,
                                              const float* __restrict__ residual,
                                              float* __restrict__ Cout,
                                              int M, int N, int K, int do_gelu) {
    (void)M;
    __shared__ __align__(16) unsigned As[BM][BK / 2];   // bf16 pairs: 64 x 64B
    __shared__ __align__(16) unsigned Bs[BN][BK / 2];
    int tid = threadIdx.x;
    int wid = tid >> 5, lane = tid & 31;
    int warpM = wid & 3, warpN = wid >> 2;              // 4 x 2 wave grid
    int m0 = blockIdx.y * BM, n0 = blockIdx.x * BN;
    int g = lane >> 4, l16 = lane & 15;
    v8f acc0 = {}; v8f acc1 = {};
    const unsigned* WbU = (const unsigned*)Wb;          // K even -> pair-indexable
    int K2 = K >> 1;

    // tile-fill assignment: row r = tid/4, quarter q = tid%4 (8 k-elems each)
    int lr = tid >> 2, lq = tid & 3;
    const float* Arow = A + (long)(m0 + lr) * K + 8 * lq;
    const unsigned* Brow = WbU + (long)(n0 + lr) * K2 + 4 * lq;
    bool brow_ok = (n0 + lr) < N;
    bool nfull = (n0 + BN) <= N;                        // block-uniform

    // fragment rows (ISA 16-bit layout: two runs of 4 consecutive uints per fragment)
    int arow = warpM * 16 + l16;
    int brow0 = warpN * 32 + l16;
    int brow1 = brow0 + 16;

    auto mma_step = [&]() {
        uint4 alo = *(const uint4*)&As[arow][g * 4];
        uint4 ahi = *(const uint4*)&As[arow][8 + g * 4];
        uint4 b0lo = *(const uint4*)&Bs[brow0][g * 4];
        uint4 b0hi = *(const uint4*)&Bs[brow0][8 + g * 4];
        uint4 b1lo = *(const uint4*)&Bs[brow1][g * 4];
        uint4 b1hi = *(const uint4*)&Bs[brow1][8 + g * 4];
        v8u au  = {alo.x, alo.y, alo.z, alo.w, ahi.x, ahi.y, ahi.z, ahi.w};
        v8u bu0 = {b0lo.x, b0lo.y, b0lo.z, b0lo.w, b0hi.x, b0hi.y, b0hi.z, b0hi.w};
        v8u bu1 = {b1lo.x, b1lo.y, b1lo.z, b1lo.w, b1hi.x, b1hi.y, b1hi.z, b1hi.w};
        v16bf a  = __builtin_bit_cast(v16bf, au);
        v16bf b0 = __builtin_bit_cast(v16bf, bu0);
        v16bf b1 = __builtin_bit_cast(v16bf, bu1);
        acc0 = __builtin_amdgcn_wmma_f32_16x16x32_bf16(false, a, false, b0, (short)0, acc0, false, false);
        acc1 = __builtin_amdgcn_wmma_f32_16x16x32_bf16(false, a, false, b1, (short)0, acc1, false, false);
    };

    // ---- main loop: full K tiles, fully unpredicated fills
    int Kmain = K & ~(BK - 1);
    for (int k0 = 0; k0 < Kmain; k0 += BK) {
        __syncthreads();
        {
            float4 f0 = *(const float4*)(Arow + k0);
            float4 f1 = *(const float4*)(Arow + k0 + 4);
            __builtin_prefetch(Arow + k0 + BK, 0, 1);   // speculative: dropped if invalid
            uint4 u;
            u.x = pack2_bf16(f0.x, f0.y);
            u.y = pack2_bf16(f0.z, f0.w);
            u.z = pack2_bf16(f1.x, f1.y);
            u.w = pack2_bf16(f1.z, f1.w);
            *(uint4*)&As[lr][4 * lq] = u;
        }
#if ASYNC_B
        if (nfull) {
            __builtin_amdgcn_global_load_async_to_lds_b128(
                (__attribute__((address_space(1))) v4i*)(unsigned long long)(Brow + (k0 >> 1)),
                (__attribute__((address_space(3))) v4i*)(unsigned)(unsigned long long)(&Bs[lr][4 * lq]),
                0, 0);
        } else
#endif
        {
            uint4 u = {0u, 0u, 0u, 0u};
            if (brow_ok) u = *(const uint4*)(Brow + (k0 >> 1));
            *(uint4*)&Bs[lr][4 * lq] = u;
        }
        wait_async0();
        __syncthreads();
        mma_step();
    }

    // ---- tail: partial K tile (K % 32, guaranteed K % 8 == 0)
    if (Kmain < K) {
        int k0 = Kmain;
        bool kq_ok = (k0 + 8 * lq) < K;                 // whole quarter in/out
        __syncthreads();
        {
            float4 f0 = {0.f, 0.f, 0.f, 0.f}, f1 = {0.f, 0.f, 0.f, 0.f};
            if (kq_ok) {
                f0 = *(const float4*)(Arow + k0);
                f1 = *(const float4*)(Arow + k0 + 4);
            }
            uint4 u;
            u.x = pack2_bf16(f0.x, f0.y);
            u.y = pack2_bf16(f0.z, f0.w);
            u.z = pack2_bf16(f1.x, f1.y);
            u.w = pack2_bf16(f1.z, f1.w);
            *(uint4*)&As[lr][4 * lq] = u;
        }
        {
            uint4 u = {0u, 0u, 0u, 0u};
            if (brow_ok && kq_ok) u = *(const uint4*)(Brow + (k0 >> 1));
            *(uint4*)&Bs[lr][4 * lq] = u;
        }
        __syncthreads();
        mma_step();
    }

#pragma unroll
    for (int r = 0; r < 8; r++) {
        int m = m0 + warpM * 16 + r + 8 * g;            // M%64==0 -> always in range
#pragma unroll
        for (int sub = 0; sub < 2; sub++) {
            int n = n0 + warpN * 32 + sub * 16 + l16;
            if (n >= N) continue;
            float v = sub ? acc1[r] : acc0[r];
            if (bias) v += bias[n];
            if (do_gelu) v = 0.5f * v * (1.f + erff(v * 0.70710678118654752f));
            if (residual) v += residual[(long)m * N + n];
            Cout[(long)m * N + n] = v;
        }
    }
}

// ---------- fused window attention: one wave per (window, head) ----------
__global__ __launch_bounds__(128) void k_attn(const float* __restrict__ QKV,
                                              const float* __restrict__ rpb,
                                              float* __restrict__ CTX,
                                              int B, int Hd, int Wd, int Cc, int hd,
                                              int shift, int total) {
    __shared__ float sP[4][16][17];
    int wid = threadIdx.x >> 5, lane = threadIdx.x & 31;
    int gid = blockIdx.x * 4 + wid;          // uniform per wave -> EXEC stays all-ones
    if (gid >= total) return;
    int head = gid % 7;
    int win  = gid / 7;
    int nW = Wd / 4;
    int wj = win % nW;
    int wi = (win / nW) % nW;
    int b  = win / (nW * nW);
    int g = lane >> 4, l16 = lane & 15;
    int C3 = 3 * Cc;
    float qscale = rsqrtf((float)hd);

    auto tok = [&](int t) -> long {
        int i = t >> 2, j = t & 3;
        int hh = wi * 4 + i + shift; if (hh >= Hd) hh -= Hd;   // roll(-shift)
        int ww = wj * 4 + j + shift; if (ww >= Wd) ww -= Wd;
        return (long)(b * Hd + hh) * Wd + ww;
    };

    // ---- scores = (q * hd^-0.5) @ k^T  (16x16, K = hd) via v_wmma_f32_16x16x4_f32
    v8f sc = {};
    {
        long tr = tok(l16);                       // lane's token: query row (A) and key col (B)
        const float* qp = QKV + tr * C3 + head * hd;
        const float* kp = qp + Cc;
        for (int kk = 0; kk < hd; kk += 4) {
            v2f a, bb;
            a.x  = qp[kk + 2 * g]     * qscale;
            a.y  = qp[kk + 2 * g + 1] * qscale;
            bb.x = kp[kk + 2 * g];
            bb.y = kp[kk + 2 * g + 1];
            sc = __builtin_amdgcn_wmma_f32_16x16x4_f32(false, a, false, bb, (short)0, sc, false, false);
        }
    }

    // ---- rel-pos bias, shifted-window mask, softmax over key dim (16 lanes / half-wave)
    float p[8];
#pragma unroll
    for (int r = 0; r < 8; r++) {
        int t = r + 8 * g;       // query token
        int s = l16;             // key token
        int di = (t >> 2) - (s >> 2) + 3;
        int dj = (t & 3)  - (s & 3)  + 3;
        float v = sc[r] + rpb[(di * 7 + dj) * 7 + head];
        if (shift > 0) {
            int ght = wi * 4 + (t >> 2), gwt = wj * 4 + (t & 3);
            int ghs = wi * 4 + (s >> 2), gws = wj * 4 + (s & 3);
            int zt = ((ght < Hd - 4) ? 0 : ((ght < Hd - 2) ? 1 : 2)) * 3 +
                     ((gwt < Wd - 4) ? 0 : ((gwt < Wd - 2) ? 1 : 2));
            int zs = ((ghs < Hd - 4) ? 0 : ((ghs < Hd - 2) ? 1 : 2)) * 3 +
                     ((gws < Wd - 4) ? 0 : ((gws < Wd - 2) ? 1 : 2));
            if (zt != zs) v -= 100.f;
        }
        p[r] = v;
    }
#pragma unroll
    for (int r = 0; r < 8; r++) {
        float mx = p[r];
        for (int o = 8; o >= 1; o >>= 1) mx = fmaxf(mx, __shfl_xor(mx, o, 16));
        float e = __expf(p[r] - mx);
        float sm = e;
        for (int o = 8; o >= 1; o >>= 1) sm += __shfl_xor(sm, o, 16);
        p[r] = e / sm;
    }
#pragma unroll
    for (int r = 0; r < 8; r++) sP[wid][r + 8 * g][l16] = p[r];
    __builtin_amdgcn_wave_barrier();

    // ---- out = P(16x16) @ V(16xhd) via f32 WMMA (K = 16 tokens)
    for (int n0b = 0; n0b < hd; n0b += 16) {
        v8f o = {};
        int dim = n0b + l16;
#pragma unroll
        for (int kk = 0; kk < 16; kk += 4) {
            v2f a, bb;
            a.x = sP[wid][l16][kk + 2 * g];
            a.y = sP[wid][l16][kk + 2 * g + 1];
            long tv0 = tok(kk + 2 * g);
            long tv1 = tok(kk + 2 * g + 1);
            bb.x = (dim < hd) ? QKV[tv0 * C3 + 2 * Cc + head * hd + dim] : 0.f;
            bb.y = (dim < hd) ? QKV[tv1 * C3 + 2 * Cc + head * hd + dim] : 0.f;
            o = __builtin_amdgcn_wmma_f32_16x16x4_f32(false, a, false, bb, (short)0, o, false, false);
        }
#pragma unroll
        for (int r = 0; r < 8; r++) {
            int t = r + 8 * g;
            if (dim < hd) CTX[tok(t) * Cc + head * hd + dim] = o[r];
        }
    }
}

// ---------- patch merge gather (2x2 -> concat 4C) ----------
__global__ void k_merge_gather(const float* __restrict__ x, float* __restrict__ out,
                               int B, int Hd, int Wd, int C) {
    long idx = (long)blockIdx.x * 256 + threadIdx.x;
    int H2 = Hd / 2, W2 = Wd / 2, C4 = 4 * C;
    long total = (long)B * H2 * W2 * C4;
    if (idx >= total) return;
    int cc = (int)(idx % C4);
    long t = idx / C4;
    int w2 = (int)(t % W2);
    int h2 = (int)((t / W2) % H2);
    int b  = (int)(t / ((long)W2 * H2));
    int q = cc / C, c = cc % C;
    int dh = (q == 1 || q == 3) ? 1 : 0;
    int dw = (q >= 2) ? 1 : 0;
    out[idx] = x[(((long)b * Hd + (2 * h2 + dh)) * Wd + (2 * w2 + dw)) * C + c];
}

// ---------- final NHWC -> NCHW flatten ----------
__global__ void k_final(const float* __restrict__ x, float* __restrict__ out, int B) {
    long idx = (long)blockIdx.x * 256 + threadIdx.x;
    long total = (long)B * 16 * 16 * 448;
    if (idx >= total) return;
    int c = (int)(idx % 448);
    long t = idx / 448;
    int w = (int)(t % 16);
    int h = (int)((t / 16) % 16);
    int b = (int)(t / 256);
    out[((long)b * 448 + c) * 256 + h * 16 + w] = x[idx];
}

// =====================================================================
extern "C" void kernel_launch(void* const* d_in, const int* in_sizes, int n_in,
                              void* d_out, int out_size, void* d_ws, size_t ws_size,
                              hipStream_t stream) {
    (void)in_sizes; (void)n_in; (void)out_size; (void)ws_size;
    const int B = 8;
    const float* x = (const float*)d_in[0];
    auto P = [&](int i) { return (const float*)d_in[i]; };

    // workspace carve (X 32MB | NRM/CTX 32MB | BIG 128MB | WB 4MB)
    char* ws = (char*)d_ws;
    float* X   = (float*)ws;
    float* NRM = (float*)(ws + ((size_t)32 << 20));
    float* CTX = NRM;                                   // alias: LN-out dead once qkv GEMM done
    float* BIG = (float*)(ws + ((size_t)64 << 20));     // qkv-out / mlp-hidden / merge-concat
    unsigned short* WB = (unsigned short*)(ws + ((size_t)192 << 20));

    auto cdiv = [](long a, long b) { return (int)((a + b - 1) / b); };
    auto cast_w = [&](const float* src, long n) {
        k_cast_bf16<<<cdiv(n, 256), 256, 0, stream>>>(src, WB, n);
    };
    auto gemm = [&](const float* A, const float* bias, const float* res, float* out,
                    long M, int N, int K, int gelu) {
        dim3 grid((N + BN - 1) / BN, (unsigned)((M + BM - 1) / BM));
        k_gemm<<<grid, 256, 0, stream>>>(A, WB, bias, res, out, (int)M, N, K, gelu);
    };

    // ---- patch embed + LN ----
    long tokens = (long)B * 128 * 128;
    k_conv<<<cdiv(tokens * 56, 256), 256, 0, stream>>>(x, P(1), P(2), BIG, B);
    k_ln<<<cdiv(tokens, 4), 128, 0, stream>>>(BIG, X, P(3), P(4), tokens, 56);

    // ---- stages ----
    int H = 128, C = 56;
    int pi = 5;
    for (int st = 0; st < 4; st++) {
        int hd = C / 7;
        tokens = (long)B * H * H;
        int nW = H / 4;
        int nwaves = B * nW * nW * 7;
        for (int blk = 0; blk < 2; blk++) {
            int shift = (blk % 2 == 0) ? 0 : 2;
            const float *n1w = P(pi), *n1b = P(pi + 1), *qkvw = P(pi + 2), *qkvb = P(pi + 3),
                        *rpb = P(pi + 4), *projw = P(pi + 5), *projb = P(pi + 6),
                        *n2w = P(pi + 7), *n2b = P(pi + 8), *fc1w = P(pi + 9), *fc1b = P(pi + 10),
                        *fc2w = P(pi + 11), *fc2b = P(pi + 12);
            pi += 13;
            // attention
            k_ln<<<cdiv(tokens, 4), 128, 0, stream>>>(X, NRM, n1w, n1b, tokens, C);
            cast_w(qkvw, (long)3 * C * C);
            gemm(NRM, qkvb, nullptr, BIG, tokens, 3 * C, C, 0);
            k_attn<<<cdiv(nwaves, 4), 128, 0, stream>>>(BIG, rpb, CTX, B, H, H, C, hd, shift, nwaves);
            cast_w(projw, (long)C * C);
            gemm(CTX, projb, X, X, tokens, C, C, 0);
            // MLP
            k_ln<<<cdiv(tokens, 4), 128, 0, stream>>>(X, NRM, n2w, n2b, tokens, C);
            cast_w(fc1w, (long)4 * C * C);
            gemm(NRM, fc1b, nullptr, BIG, tokens, 4 * C, C, 1);
            cast_w(fc2w, (long)4 * C * C);
            gemm(BIG, fc2b, X, X, tokens, C, 4 * C, 0);
        }
        if (st < 3) {
            const float *nw = P(pi), *nb = P(pi + 1), *rw = P(pi + 2);
            pi += 3;
            long tok2 = tokens / 4;
            k_merge_gather<<<cdiv(tok2 * 4 * C, 256), 256, 0, stream>>>(X, BIG, B, H, H, C);
            k_ln<<<cdiv(tok2, 4), 128, 0, stream>>>(BIG, NRM, nw, nb, tok2, 4 * C);
            cast_w(rw, (long)2 * C * 4 * C);
            gemm(NRM, nullptr, nullptr, X, tok2, 2 * C, 4 * C, 0);
            H /= 2;
            C *= 2;
        }
    }

    // ---- final reshape ----
    k_final<<<cdiv((long)B * 256 * 448, 256), 256, 0, stream>>>(X, (float*)d_out, B);
}
